// GraphSummary_7215545057977
// MI455X (gfx1250) — matrix operand order repaired and checked by reading, not compiled, not run on hardware
//
#include <hip/hip_runtime.h>
#include <hip/hip_bf16.h>

typedef __attribute__((ext_vector_type(16))) _Float16 v16h;
typedef __attribute__((ext_vector_type(8)))  _Float16 h8;
typedef __attribute__((ext_vector_type(4)))  _Float16 h4;
typedef __attribute__((ext_vector_type(8)))  float    v8f;

#define BATCH   32
#define NNODES  8192
#define DIM     128
#define HID     128
#define WT_PITCH 136   // halfs; 272B rows -> 4-bank shift per row, 16B aligned
#define X_PITCH  136
#define WPB 4                       // waves per block
#define NODES_PER_BLOCK (WPB * 16)  // 64 nodes per block

__device__ __forceinline__ float silu_f(float v) {
    return v / (1.0f + __expf(-v));
}

// A fragment: 16x32 f16 tile, lane l(0-15)=row M=l, halves: lanes0-15 K {0..7,16..23},
// lanes16-31 K {8..15,24..31} (pairs per VGPR per ISA table)
__device__ __forceinline__ v16h load_a_frag(const _Float16* tile, int l, int half, int kc) {
    const _Float16* p = tile + l * X_PITCH + kc * 32 + half * 8;
    v16h a;
    ((h8*)&a)[0] = *(const h8*)(p);        // K = kbase + half*8 + 0..7
    ((h8*)&a)[1] = *(const h8*)(p + 16);   // K = kbase + 16 + half*8 + 0..7
    return a;
}

// B fragment: 32x16 f16 (KxN), lane l = col N; lanes0-15 hold K {0..7,16..23}, lanes16-31 K {8..15,24..31}
__device__ __forceinline__ v16h load_b_frag(const _Float16* wt, int col, int half, int kc) {
    const _Float16* p = wt + col * WT_PITCH + kc * 32 + half * 8;
    v16h b;
    ((h8*)&b)[0] = *(const h8*)(p);
    ((h8*)&b)[1] = *(const h8*)(p + 16);
    return b;
}

// ---------------- Kernel 1: per-node gate scores ----------------
__global__ __launch_bounds__(128)
void gate_score_kernel(const float* __restrict__ x,
                       const float* __restrict__ Wg1,
                       const float* __restrict__ bg1,
                       const float* __restrict__ Wg2,
                       float* __restrict__ scores) {
    __shared__ _Float16 sWt[HID * WT_PITCH];       // Wg1 transposed: [h][d]
    __shared__ _Float16 sX[WPB][16 * X_PITCH];     // per-wave x tile f16
    __shared__ float    sB1[HID];
    __shared__ float    sW2[HID];

    const int tid  = threadIdx.x;
    const int wv   = tid >> 5;
    const int lane = tid & 31;
    const int l    = lane & 15;
    const int half = lane >> 4;
    const int node0 = blockIdx.x * NODES_PER_BLOCK + wv * 16;

    // stage Wg1 transposed (global coalesced read, f32->f16)
    for (int i = tid; i < DIM * HID; i += 128) {
        int d = i >> 7, h = i & 127;
        sWt[h * WT_PITCH + d] = (_Float16)Wg1[i];
    }
    if (tid < HID) { sB1[tid] = bg1[tid]; sW2[tid] = Wg2[tid]; }

    // stage x tile (16 rows x 128 f32) as f16
    const float4* xg = (const float4*)(x + (size_t)node0 * DIM);
    #pragma unroll
    for (int j = 0; j < 16; ++j) {
        int li = lane + 32 * j;                 // 0..511 float4s
        float4 v = xg[li];
        int pos = li * 4, row = pos >> 7, col = pos & 127;
        h4 hv = { (_Float16)v.x, (_Float16)v.y, (_Float16)v.z, (_Float16)v.w };
        *(h4*)&sX[wv][row * X_PITCH + col] = hv;
    }
    __syncthreads();

    v8f acc[8];
    #pragma unroll
    for (int nt = 0; nt < 8; ++nt) acc[nt] = v8f{0.f,0.f,0.f,0.f,0.f,0.f,0.f,0.f};

    #pragma unroll
    for (int kc = 0; kc < 4; ++kc) {
        v16h a = load_a_frag(&sX[wv][0], l, half, kc);
        #pragma unroll
        for (int nt = 0; nt < 8; ++nt) {
            v16h b = load_b_frag(sWt, nt * 16 + l, half, kc);
            acc[nt] = __builtin_amdgcn_wmma_f32_16x16x32_f16(
                false, a, false, b, (short)0, acc[nt], false, false);
        }
    }

    // SiLU + dot with Wg2, reduce over hidden dim
    float partial[8];
    #pragma unroll
    for (int r = 0; r < 8; ++r) partial[r] = 0.f;
    #pragma unroll
    for (int nt = 0; nt < 8; ++nt) {
        int h = nt * 16 + l;
        float bb = sB1[h], w2 = sW2[h];
        #pragma unroll
        for (int r = 0; r < 8; ++r) {
            float v = acc[nt][r] + bb;
            partial[r] += silu_f(v) * w2;
        }
    }
    // reduce across the 16 lanes of each half-wave (each lane covered 8 h-cols)
    #pragma unroll
    for (int r = 0; r < 8; ++r) {
        #pragma unroll
        for (int m = 1; m < 16; m <<= 1)
            partial[r] += __shfl_xor(partial[r], m, 16);
    }
    if (l == 0) {
        #pragma unroll
        for (int r = 0; r < 8; ++r)
            scores[(size_t)node0 + half * 8 + r] = partial[r];
    }
}

// ---------------- Kernel 2: per-batch softmax stats (max, sumexp) ----------------
__global__ __launch_bounds__(256)
void softmax_stats_kernel(const float* __restrict__ scores, float* __restrict__ stats) {
    __shared__ float red[256];
    const int b = blockIdx.x, tid = threadIdx.x;
    const float* g = scores + (size_t)b * NNODES;

    float m = -3.402823466e38f;
    for (int i = tid; i < NNODES; i += 256) m = fmaxf(m, g[i]);
    red[tid] = m; __syncthreads();
    for (int s = 128; s > 0; s >>= 1) {
        if (tid < s) red[tid] = fmaxf(red[tid], red[tid + s]);
        __syncthreads();
    }
    m = red[0]; __syncthreads();

    float ssum = 0.f;
    for (int i = tid; i < NNODES; i += 256) ssum += __expf(g[i] - m);
    red[tid] = ssum; __syncthreads();
    for (int s = 128; s > 0; s >>= 1) {
        if (tid < s) red[tid] += red[tid + s];
        __syncthreads();
    }
    if (tid == 0) { stats[2 * b] = m; stats[2 * b + 1] = red[0]; }
}

// ---------------- Kernel 3: weighted SiLU(x@Wn1+bn1) accumulation ----------------
__global__ __launch_bounds__(128)
void pool_kernel(const float* __restrict__ x,
                 const float* __restrict__ Wn1,
                 const float* __restrict__ bn1,
                 const float* __restrict__ scores,
                 const float* __restrict__ stats,
                 float* __restrict__ sacc) {
    __shared__ _Float16 sWt[HID * WT_PITCH];
    __shared__ _Float16 sX[WPB][16 * X_PITCH];
    __shared__ float    sB1[HID];
    __shared__ float    sblk[HID];

    const int tid  = threadIdx.x;
    const int wv   = tid >> 5;
    const int lane = tid & 31;
    const int l    = lane & 15;
    const int half = lane >> 4;
    const int node0 = blockIdx.x * NODES_PER_BLOCK + wv * 16;
    const int b = node0 >> 13;  // /8192; block never straddles a batch

    for (int i = tid; i < DIM * HID; i += 128) {
        int d = i >> 7, h = i & 127;
        sWt[h * WT_PITCH + d] = (_Float16)Wn1[i];
    }
    if (tid < HID) { sB1[tid] = bn1[tid]; sblk[tid] = 0.f; }

    const float4* xg = (const float4*)(x + (size_t)node0 * DIM);
    #pragma unroll
    for (int j = 0; j < 16; ++j) {
        int li = lane + 32 * j;
        float4 v = xg[li];
        int pos = li * 4, row = pos >> 7, col = pos & 127;
        h4 hv = { (_Float16)v.x, (_Float16)v.y, (_Float16)v.z, (_Float16)v.w };
        *(h4*)&sX[wv][row * X_PITCH + col] = hv;
    }
    __syncthreads();

    v8f acc[8];
    #pragma unroll
    for (int nt = 0; nt < 8; ++nt) acc[nt] = v8f{0.f,0.f,0.f,0.f,0.f,0.f,0.f,0.f};

    #pragma unroll
    for (int kc = 0; kc < 4; ++kc) {
        v16h a = load_a_frag(&sX[wv][0], l, half, kc);
        #pragma unroll
        for (int nt = 0; nt < 8; ++nt) {
            v16h b2 = load_b_frag(sWt, nt * 16 + l, half, kc);
            acc[nt] = __builtin_amdgcn_wmma_f32_16x16x32_f16(
                false, a, false, b2, (short)0, acc[nt], false, false);
        }
    }

    // softmax weights for the 8 rows this lane's C registers cover
    const float maxb = stats[2 * b];
    const float invs = 1.0f / stats[2 * b + 1];
    float w[8];
    #pragma unroll
    for (int r = 0; r < 8; ++r) {
        float g = scores[(size_t)node0 + half * 8 + r];
        w[r] = __expf(g - maxb) * invs;
    }

    #pragma unroll
    for (int nt = 0; nt < 8; ++nt) {
        int h = nt * 16 + l;
        float bb = sB1[h];
        float ha = 0.f;
        #pragma unroll
        for (int r = 0; r < 8; ++r) {
            float v = acc[nt][r] + bb;
            ha += w[r] * silu_f(v);
        }
        ha += __shfl_xor(ha, 16, 32);   // combine rows 0-7 with rows 8-15 halves
        if (half == 0) atomicAdd(&sblk[h], ha);
    }
    __syncthreads();
    atomicAdd(&sacc[b * HID + tid], sblk[tid]);   // 128 threads, 128 h-slots
}

// ---------------- Kernel 4: out = sacc @ Wn2 + bn2 ----------------
__global__ __launch_bounds__(128)
void finalize_kernel(const float* __restrict__ sacc,
                     const float* __restrict__ Wn2,
                     const float* __restrict__ bn2,
                     float* __restrict__ out) {
    const int b = blockIdx.x, h = threadIdx.x;
    const float* sv = sacc + b * HID;
    float a = bn2[h];
    for (int k = 0; k < HID; ++k) a = fmaf(sv[k], Wn2[k * HID + h], a);
    out[b * HID + h] = a;
}

extern "C" void kernel_launch(void* const* d_in, const int* in_sizes, int n_in,
                              void* d_out, int out_size, void* d_ws, size_t ws_size,
                              hipStream_t stream) {
    const float* x   = (const float*)d_in[0];
    const float* Wg1 = (const float*)d_in[1];
    const float* bg1 = (const float*)d_in[2];
    const float* Wg2 = (const float*)d_in[3];
    // d_in[4] = bg2: constant shift, cancels in softmax — unused
    const float* Wn1 = (const float*)d_in[5];
    const float* bn1 = (const float*)d_in[6];
    const float* Wn2 = (const float*)d_in[7];
    const float* bn2 = (const float*)d_in[8];
    float* out = (float*)d_out;

    float* scores = (float*)d_ws;                       // B*N floats (1 MB)
    float* stats  = scores + (size_t)BATCH * NNODES;    // 2*B floats
    float* sacc   = stats + 2 * BATCH;                  // B*H floats

    const int total_nodes = BATCH * NNODES;
    const int nblocks = total_nodes / NODES_PER_BLOCK;  // 4096

    gate_score_kernel<<<nblocks, 128, 0, stream>>>(x, Wg1, bg1, Wg2, scores);
    softmax_stats_kernel<<<BATCH, 256, 0, stream>>>(scores, stats);
    hipMemsetAsync(sacc, 0, (size_t)BATCH * HID * sizeof(float), stream);
    pool_kernel<<<nblocks, 128, 0, stream>>>(x, Wn1, bn1, scores, stats, sacc);
    finalize_kernel<<<BATCH, 128, 0, stream>>>(sacc, Wn2, bn2, out);
}
